// QuantizedExpertFusedCPLRPLModel_43868795961853
// MI455X (gfx1250) — matrix operand order repaired and checked by reading, not compiled, not run on hardware
//
#include <hip/hip_runtime.h>

typedef __attribute__((ext_vector_type(16))) _Float16 v16h;
typedef __attribute__((ext_vector_type(8)))  _Float16 v8h;
typedef __attribute__((ext_vector_type(8)))  float    v8f;
typedef __attribute__((ext_vector_type(4)))  float    v4f;
typedef __attribute__((ext_vector_type(4)))  int      v4i;

namespace {
constexpr int E = 8, T = 4096, H = 2048;
constexpr int BM = 256, BN = 128, BK = 32;
constexpr int THREADS = 256;            // 8 wave32s; waves tiled 4(M) x 2(N), 64x64 each
}

union V16 { v16h v; v8h h[2]; };
union H2  { _Float16 h[2]; unsigned u; };

// LAYER1: act = f32 x [E,T,H], out = f16 h (workspace)
// !LAYER1: act = f16 h,        out = f32 out
template<bool LAYER1>
__global__ __launch_bounds__(THREADS)
void moe_wmma_gemm(const void* __restrict__ act,
                   const int*  __restrict__ wq,
                   const float* __restrict__ scale,
                   void* __restrict__ outp)
{
    // double-buffered tiles: A [m][k] 2x16KB, B transposed [n][k] 2x8KB
    __shared__ __attribute__((aligned(32))) _Float16 sA[2][BM * BK];
    __shared__ __attribute__((aligned(32))) _Float16 sB[2][BN * BK];

    const int e  = blockIdx.z;
    const int m0 = blockIdx.y * BM;
    const int n0 = blockIdx.x * BN;

    const int tid  = threadIdx.x;
    const int lane = tid & 31;
    const int wave = tid >> 5;
    const int wm   = wave >> 1;      // 0..3 : 64-row band
    const int wn   = wave & 1;       // 0..1 : 64-col band
    const int l15  = lane & 15;
    const int lhi  = lane >> 4;      // half-wave K select per ISA layout

    const int*      w   = wq    + (size_t)e * H * H;
    const float*    sc  = scale + (size_t)e * H;
    const float*    a32 = (const float*)act    + (size_t)e * T * H;
    const _Float16* a16 = (const _Float16*)act + (size_t)e * T * H;

    // staging assignments: A -> thread tid stages row (m0+tid), 32 k-values
    //                      B -> thread stages k-pair (kk,kk+1) x 8 n-values, packed b32
    const int kk = (tid & 15) * 2;       // 0..30 (even)
    const int nn = (tid >> 4) * 8;       // 0..120

    v8h      aReg[4];                     // 32 halves of A staged in registers
    unsigned bReg[8];                     // 8 packed f16-pairs of B^T

    auto stageLoad = [&](int k0) {
        if constexpr (LAYER1) {
            const float* src = a32 + (size_t)(m0 + tid) * H + k0;
            #pragma unroll
            for (int q = 0; q < 4; ++q) {
                v4f f0 = *(const v4f*)(src + q * 8);
                v4f f1 = *(const v4f*)(src + q * 8 + 4);
                v8h o;
                #pragma unroll
                for (int t = 0; t < 4; ++t) { o[t] = (_Float16)f0[t]; o[4 + t] = (_Float16)f1[t]; }
                aReg[q] = o;
            }
        } else {
            const _Float16* src = a16 + (size_t)(m0 + tid) * H + k0;
            #pragma unroll
            for (int q = 0; q < 4; ++q) aReg[q] = *(const v8h*)(src + q * 8);
        }
        const int* s0 = w + (size_t)(k0 + kk) * H + n0 + nn;       // row k
        const int* s1 = s0 + H;                                     // row k+1
        v4i q0 = *(const v4i*)(s0);
        v4i q1 = *(const v4i*)(s0 + 4);
        v4i q2 = *(const v4i*)(s1);
        v4i q3 = *(const v4i*)(s1 + 4);
        #pragma unroll
        for (int i = 0; i < 4; ++i) {
            H2 p;
            p.h[0] = (_Float16)(float)q0[i];  p.h[1] = (_Float16)(float)q2[i];
            bReg[i] = p.u;
            p.h[0] = (_Float16)(float)q1[i];  p.h[1] = (_Float16)(float)q3[i];
            bReg[4 + i] = p.u;
        }
    };

    auto stageStore = [&](int buf) {
        #pragma unroll
        for (int q = 0; q < 4; ++q)
            *(v8h*)&sA[buf][tid * BK + q * 8] = aReg[q];
        #pragma unroll
        for (int i = 0; i < 8; ++i)
            *(unsigned*)&sB[buf][(nn + i) * BK + kk] = bReg[i];   // [n][k], k-pair packed
    };

    v8f acc[4][4] = {};

    stageLoad(0);
    stageStore(0);
    __syncthreads();

    int cur = 0;
    for (int k0 = 0; k0 < H; k0 += BK) {
        const bool more = (k0 + BK) < H;
        if (more) {
            stageLoad(k0 + BK);                       // HBM loads fly during WMMA below
            __builtin_prefetch(w + (size_t)(k0 + 2 * BK + kk) * H + n0 + nn, 0, 1);
        }

        // ---- fragments per documented 16-bit WMMA VGPR layouts ----
        V16 af[4], bf[4];
        #pragma unroll
        for (int i = 0; i < 4; ++i) {
            // A 16x32: lanes 0-15 row M with K 0..7 & 16..23; lanes 16-31 K 8..15 & 24..31
            const _Float16* p = &sA[cur][(wm * 64 + i * 16 + l15) * BK + lhi * 8];
            af[i].h[0] = *(const v8h*)(p);
            af[i].h[1] = *(const v8h*)(p + 16);
        }
        #pragma unroll
        for (int j = 0; j < 4; ++j) {
            // B 32x16: lanes 0-15 col N with K 0..7 & 16..23; lanes 16-31 K 8..15 & 24..31
            const _Float16* p = &sB[cur][(wn * 64 + j * 16 + l15) * BK + lhi * 8];
            bf[j].h[0] = *(const v8h*)(p);
            bf[j].h[1] = *(const v8h*)(p + 16);
        }

        #pragma unroll
        for (int i = 0; i < 4; ++i)
            #pragma unroll
            for (int j = 0; j < 4; ++j)
                acc[i][j] = __builtin_amdgcn_wmma_f32_16x16x32_f16(
                    false, af[i].v, false, bf[j].v,
                    (short)0, acc[i][j], false, false);

        if (more) stageStore(cur ^ 1);                // write the *other* buffer
        __syncthreads();                              // one barrier per K-iteration
        cur ^= 1;
    }

    // ---- epilogue: deferred per-output-channel dequant, then store ----
    #pragma unroll
    for (int j = 0; j < 4; ++j) {
        const int col = n0 + wn * 64 + j * 16 + l15;
        const float s = sc[col];
        #pragma unroll
        for (int i = 0; i < 4; ++i) {
            const int rowBase = m0 + wm * 64 + i * 16 + lhi * 8;
            #pragma unroll
            for (int r = 0; r < 8; ++r) {
                const float v = acc[i][j][r] * s;
                const size_t idx = (size_t)e * T * H + (size_t)(rowBase + r) * H + col;
                if constexpr (LAYER1) ((_Float16*)outp)[idx] = (_Float16)v;
                else                  ((float*)outp)[idx]    = v;
            }
        }
    }
}

extern "C" void kernel_launch(void* const* d_in, const int* in_sizes, int n_in,
                              void* d_out, int out_size, void* d_ws, size_t ws_size,
                              hipStream_t stream)
{
    (void)in_sizes; (void)n_in; (void)out_size; (void)ws_size;
    const float* x   = (const float*)d_in[0];
    const int*   w1q = (const int*)d_in[1];
    const float* w1s = (const float*)d_in[2];
    const int*   w2q = (const int*)d_in[3];
    const float* w2s = (const float*)d_in[4];

    // f16 intermediate h: E*T*H * 2B = 128 MiB in workspace
    _Float16* hbuf = (_Float16*)d_ws;

    dim3 grid(H / BN, T / BM, E);   // (16, 16, 8)
    dim3 block(THREADS);
    moe_wmma_gemm<true ><<<grid, block, 0, stream>>>(x,    w1q, w1s, hbuf);
    moe_wmma_gemm<false><<<grid, block, 0, stream>>>(hbuf, w2q, w2s, d_out);
}
